// MultiheadCrossAttention_53085795779465
// MI455X (gfx1250) — matrix operand order repaired and checked by reading, not compiled, not run on hardware
//
#include <hip/hip_runtime.h>
#include <stdint.h>

// ---------------------------------------------------------------------------
// MultiheadCrossAttention on gfx1250 (CDNA5, wave32, WMMA 16x16x32 f16)
//   B=4, Lq=512, Lk=2048, EMBED=ATTN=1024, HEADS=16, HEAD_DIM=64
// f16 compute / f32 accumulate; async global->LDS (ASYNCcnt) double-buffered
// staging feeds the WMMA pipelines in both the GEMMs and flash attention.
// ---------------------------------------------------------------------------

typedef __attribute__((ext_vector_type(16))) _Float16 v16h;
typedef __attribute__((ext_vector_type(8)))  _Float16 v8h;
typedef __attribute__((ext_vector_type(8)))  float    v8f;
typedef __attribute__((ext_vector_type(4)))  int      v4i;

typedef __attribute__((address_space(1))) v4i* gptr_v4i;
typedef __attribute__((address_space(3))) v4i* lptr_v4i;

#define DEVINL __device__ __forceinline__

// ---- WMMA wrapper ----------------------------------------------------------
DEVINL v8f wmma_f32_16x16x32(v16h a, v16h b, v8f c) {
  return __builtin_amdgcn_wmma_f32_16x16x32_f16(
      /*neg_a=*/false, a, /*neg_b=*/false, b,
      /*c_mod=*/(short)0, c, /*reuse_a=*/false, /*reuse_b=*/false);
}

// ---- async global -> LDS, 16B per lane (GLOBAL_LOAD_ASYNC_TO_LDS_B128) -----
DEVINL void async_copy_b128(const _Float16* g, const _Float16* l) {
#if __has_builtin(__builtin_amdgcn_global_load_async_to_lds_b128)
  __builtin_amdgcn_global_load_async_to_lds_b128(
      (gptr_v4i)(uintptr_t)g, (lptr_v4i)(uintptr_t)l, 0, 0);
#else
  asm volatile("global_load_async_to_lds_b128 %0, %1, off"
               :: "v"((uint32_t)(uintptr_t)l), "v"((uint64_t)(uintptr_t)g)
               : "memory");
#endif
}

DEVINL void wait_asynccnt0() {
#if __has_builtin(__builtin_amdgcn_s_wait_asynccnt)
  __builtin_amdgcn_s_wait_asynccnt(0);
#else
  asm volatile("s_wait_asynccnt 0" ::: "memory");
#endif
}

// ---- A fragment: 16(M) x 32(K) from row-major f16, stride lda (halves) -----
DEVINL v16h load_a_frag(const _Float16* p, int lda) {
  const int lane = threadIdx.x & 31;
  const int m  = lane & 15;
  const int hs = lane >> 4;
  const _Float16* row = p + (long)m * lda + hs * 8;
  v8h lo = *(const v8h*)(row);        // K = hs*8 .. +7
  v8h hi = *(const v8h*)(row + 16);   // K = 16+hs*8 .. +7
  return __builtin_shufflevector(lo, hi, 0,1,2,3,4,5,6,7,8,9,10,11,12,13,14,15);
}

// ---- B fragment: 32(K) x 16(N) from N-major BT[N][K], stride ldb -----------
DEVINL v16h load_b_frag(const _Float16* p, int ldb) {
  const int lane  = threadIdx.x & 31;
  const int n     = lane & 15;
  const int khalf = lane >> 4;
  const _Float16* row = p + (long)n * ldb + khalf * 8;
  v8h lo = *(const v8h*)(row);        // K = khalf*8 .. +7
  v8h hi = *(const v8h*)(row + 16);   // K = 16+khalf*8 .. +7
  return __builtin_shufflevector(lo, hi, 0,1,2,3,4,5,6,7,8,9,10,11,12,13,14,15);
}

// ---------------------------------------------------------------------------
// f32 -> f16 elementwise convert
// ---------------------------------------------------------------------------
__global__ void cvt_f32_to_f16_kernel(const float* __restrict__ src,
                                      _Float16* __restrict__ dst, long n) {
  long i = (long)blockIdx.x * blockDim.x + threadIdx.x;
  long stride = (long)gridDim.x * blockDim.x;
  for (; i < n; i += stride) dst[i] = (_Float16)src[i];
}

// ---------------------------------------------------------------------------
// f32 [R][C] -> f16 transposed [C][R]
// ---------------------------------------------------------------------------
__global__ void transpose_cvt_kernel(const float* __restrict__ src,
                                     _Float16* __restrict__ dst, int R, int C) {
  __shared__ float tile[32][33];
  int c0 = blockIdx.x * 32, r0 = blockIdx.y * 32;
  for (int i = threadIdx.y; i < 32; i += 8) {
    int r = r0 + i, c = c0 + threadIdx.x;
    tile[i][threadIdx.x] = (r < R && c < C) ? src[(long)r * C + c] : 0.0f;
  }
  __syncthreads();
  for (int i = threadIdx.y; i < 32; i += 8) {
    int c = c0 + i, r = r0 + threadIdx.x;
    if (c < C && r < R) dst[(long)c * R + r] = (_Float16)tile[threadIdx.x][i];
  }
}

// ---------------------------------------------------------------------------
// V columns of kv, transposed per batch: vT[b][d][t]
// ---------------------------------------------------------------------------
__global__ void transpose_v_kernel(const _Float16* __restrict__ kv,
                                   _Float16* __restrict__ vT) {
  __shared__ _Float16 tile[32][33];
  int t0 = blockIdx.x * 32, d0 = blockIdx.y * 32, b = blockIdx.z;
  for (int i = threadIdx.y; i < 32; i += 8)
    tile[i][threadIdx.x] =
        kv[((long)(b * 2048 + t0 + i)) * 2048 + 1024 + d0 + threadIdx.x];
  __syncthreads();
  for (int i = threadIdx.y; i < 32; i += 8)
    vT[(long)b * 1024 * 2048 + (long)(d0 + i) * 2048 + t0 + threadIdx.x] =
        tile[threadIdx.x][i];
}

// ---------------------------------------------------------------------------
// WMMA GEMM:  C[M,N] = A[M,K] @ BT[N,K]^T + bias
// 256 threads = 8 waves (4M x 2N), block tile 128x128, K-step 32.
// A/B tiles async-staged into LDS (double buffered, 80B-padded rows).
// cmode: 0 -> f32 out, 1 -> f16 out
// ---------------------------------------------------------------------------
#define LDA_S 40   // 32 halves + 8 pad (80 B rows: 16B aligned, bank-friendly)

__global__ __launch_bounds__(256)
void gemm_wmma_kernel(const _Float16* __restrict__ A, int lda,
                      const _Float16* __restrict__ BT, int ldb,
                      const float* __restrict__ bias,
                      void* __restrict__ Cout, int ldc, int Kd, int cmode) {
  __shared__ _Float16 Asm[2][128 * LDA_S];   // 2 x 10 KB
  __shared__ _Float16 Bsm[2][128 * LDA_S];   // 2 x 10 KB
  const int tid  = threadIdx.x;
  const int wave = tid >> 5, lane = tid & 31;
  const int nl = lane & 15, hs = lane >> 4;
  const int wm = wave & 3, wn = wave >> 2;
  const long m_blk = (long)blockIdx.y * 128;
  const long n_blk = (long)blockIdx.x * 128;
  const _Float16* Ab = A + m_blk * lda;
  const _Float16* Bb = BT + n_blk * ldb;

  // cooperative stage of one 128x32 f16 tile: 512 b128 chunks, 2 per thread
  auto stage = [&](const _Float16* src, int ld, _Float16* dst, int k0) {
#pragma unroll
    for (int s = 0; s < 2; ++s) {
      const int id = tid + 256 * s;          // 0..511
      const int row = id >> 2, ch = id & 3;  // 128 rows x 4 chunks of 8 halves
      async_copy_b128(src + (long)row * ld + k0 + ch * 8,
                      dst + row * LDA_S + ch * 8);
    }
  };

  stage(Ab, lda, &Asm[0][0], 0);
  stage(Bb, ldb, &Bsm[0][0], 0);

  v8f acc[2][4] = {};
  int cur = 0;
  for (int k0 = 0; k0 < Kd; k0 += 32, cur ^= 1) {
    wait_asynccnt0();
    __syncthreads();
    if (k0 + 32 < Kd) {
      stage(Ab, lda, &Asm[cur ^ 1][0], k0 + 32);
      stage(Bb, ldb, &Bsm[cur ^ 1][0], k0 + 32);
    }
    v16h a0 = load_a_frag(&Asm[cur][(wm * 32) * LDA_S], LDA_S);
    v16h a1 = load_a_frag(&Asm[cur][(wm * 32 + 16) * LDA_S], LDA_S);
#pragma unroll
    for (int j = 0; j < 4; ++j) {
      v16h bj = load_b_frag(&Bsm[cur][(wn * 64 + j * 16) * LDA_S], LDA_S);
      acc[0][j] = wmma_f32_16x16x32(a0, bj, acc[0][j]);
      acc[1][j] = wmma_f32_16x16x32(a1, bj, acc[1][j]);
    }
  }
  // epilogue: C layout -> row r+8*hs, col nl within each 16x16
#pragma unroll
  for (int j = 0; j < 4; ++j) {
    const long n = n_blk + wn * 64 + j * 16 + nl;
    const float bv = bias ? bias[n] : 0.0f;
#pragma unroll
    for (int i = 0; i < 2; ++i)
#pragma unroll
      for (int r = 0; r < 8; ++r) {
        const long m = m_blk + wm * 32 + i * 16 + r + 8 * hs;
        const float v = acc[i][j][r] + bv;
        if (cmode == 0) ((float*)Cout)[m * ldc + n] = v;
        else            ((_Float16*)Cout)[m * ldc + n] = (_Float16)v;
      }
  }
}

// ---------------------------------------------------------------------------
// Flash attention. grid = (Lq/128, HEADS, B), block = 256 (8 waves).
// K/V tiles (64x64) async-staged once per block (waves share the stream),
// per-wave online softmax, P relayout through per-wave LDS tile.
// ---------------------------------------------------------------------------
#define LDK_S 80   // 64 halves + 16 pad (160 B rows)

__global__ __launch_bounds__(256)
void flash_attn_wmma_kernel(const _Float16* __restrict__ q,
                            const _Float16* __restrict__ kbuf,
                            const _Float16* __restrict__ vT,
                            _Float16* __restrict__ xbuf) {
  constexpr int LK = 2048, LQ = 512;
  __shared__ _Float16 Ksm[2][64 * LDK_S];   // 2 x 10 KB
  __shared__ _Float16 Vsm[2][64 * LDK_S];   // 2 x 10 KB
  __shared__ _Float16 Plds[8][16 * 64];     // 16 KB (per-wave P staging)

  const int tid  = threadIdx.x;
  const int wave = tid >> 5, lane = tid & 31;
  const int nl = lane & 15, hs = lane >> 4;
  const int h = blockIdx.y, b = blockIdx.z;
  const int m0 = blockIdx.x * 128 + wave * 16;
  _Float16* Pw = &Plds[wave][0];

  const _Float16* Qp = q + ((long)(b * LQ + m0)) * 1024 + h * 64;
  const v16h qa0 = load_a_frag(Qp, 1024);        // d 0..31
  const v16h qa1 = load_a_frag(Qp + 32, 1024);   // d 32..63

  const _Float16* Kbase = kbuf + ((long)b * LK) * 2048 + h * 64;  // rows: tokens
  const _Float16* Vbase = vT + ((long)b * 1024 + h * 64) * LK;    // rows: d

  // cooperative stage of one 64x64 f16 tile: 512 b128 chunks, 2 per thread
  auto stageK = [&](int kt, int buf) {
#pragma unroll
    for (int s = 0; s < 2; ++s) {
      const int id = tid + 256 * s;
      const int row = id >> 3, ch = id & 7;    // 64 rows x 8 chunks of 8 halves
      async_copy_b128(Kbase + (long)(kt + row) * 2048 + ch * 8,
                      &Ksm[buf][row * LDK_S + ch * 8]);
    }
  };
  auto stageV = [&](int kt, int buf) {
#pragma unroll
    for (int s = 0; s < 2; ++s) {
      const int id = tid + 256 * s;
      const int row = id >> 3, ch = id & 7;
      async_copy_b128(Vbase + (long)row * 2048 + kt + ch * 8,
                      &Vsm[buf][row * LDK_S + ch * 8]);
    }
  };

  stageK(0, 0);
  stageV(0, 0);

  v8f o[4] = {};
  float rowm[8], rowl[8];
#pragma unroll
  for (int r = 0; r < 8; ++r) { rowm[r] = -1e30f; rowl[r] = 0.0f; }

  int cur = 0;
  for (int kt = 0; kt < LK; kt += 64, cur ^= 1) {
    wait_asynccnt0();
    __syncthreads();
    if (kt + 64 < LK) { stageK(kt + 64, cur ^ 1); stageV(kt + 64, cur ^ 1); }

    // ---- S tile = Q (16x64) x K^T (64x64), from LDS ----
    v8f sc[4] = {};
#pragma unroll
    for (int j = 0; j < 4; ++j) {
      const _Float16* Kp = &Ksm[cur][(j * 16) * LDK_S];
      sc[j] = wmma_f32_16x16x32(qa0, load_b_frag(Kp, LDK_S), sc[j]);
      sc[j] = wmma_f32_16x16x32(qa1, load_b_frag(Kp + 32, LDK_S), sc[j]);
    }
    // ---- online softmax over 64 tile cols ----
    float p[4][8];
#pragma unroll
    for (int r = 0; r < 8; ++r) {
      float cm = -1e30f;
#pragma unroll
      for (int j = 0; j < 4; ++j) { sc[j][r] *= 0.125f; cm = fmaxf(cm, sc[j][r]); }
#pragma unroll
      for (int off = 1; off < 16; off <<= 1) cm = fmaxf(cm, __shfl_xor(cm, off, 32));
      const float nm = fmaxf(rowm[r], cm);
      const float corr = __expf(rowm[r] - nm);
      rowm[r] = nm;
      float ps = 0.0f;
#pragma unroll
      for (int j = 0; j < 4; ++j) { p[j][r] = __expf(sc[j][r] - nm); ps += p[j][r]; }
#pragma unroll
      for (int off = 1; off < 16; off <<= 1) ps += __shfl_xor(ps, off, 32);
      rowl[r] = rowl[r] * corr + ps;
#pragma unroll
      for (int j = 0; j < 4; ++j) o[j][r] *= corr;
    }
    // ---- C-layout P -> row-major f16 in per-wave LDS, reload as A-frags ----
#pragma unroll
    for (int j = 0; j < 4; ++j)
#pragma unroll
      for (int r = 0; r < 8; ++r)
        Pw[(r + 8 * hs) * 64 + j * 16 + nl] = (_Float16)p[j][r];
    __builtin_amdgcn_wave_barrier();
    asm volatile("s_wait_dscnt 0" ::: "memory");
    const v16h pa0 = load_a_frag(Pw, 64);
    const v16h pa1 = load_a_frag(Pw + 32, 64);
    // ---- O += P (16x64) x V (64x64), from LDS ----
#pragma unroll
    for (int j = 0; j < 4; ++j) {
      const _Float16* Vp = &Vsm[cur][(j * 16) * LDK_S];
      o[j] = wmma_f32_16x16x32(pa0, load_b_frag(Vp, LDK_S), o[j]);
      o[j] = wmma_f32_16x16x32(pa1, load_b_frag(Vp + 32, LDK_S), o[j]);
    }
    __builtin_amdgcn_wave_barrier();
    asm volatile("s_wait_dscnt 0" ::: "memory");  // P reads done before overwrite
  }
  // ---- normalize + store x (f16) ----
#pragma unroll
  for (int r = 0; r < 8; ++r) {
    const float inv = 1.0f / rowl[r];
    const long m = (long)b * LQ + m0 + r + 8 * hs;
#pragma unroll
    for (int j = 0; j < 4; ++j)
      xbuf[m * 1024 + h * 64 + j * 16 + nl] = (_Float16)(o[j][r] * inv);
  }
}

// ---------------------------------------------------------------------------
extern "C" void kernel_launch(void* const* d_in, const int* in_sizes, int n_in,
                              void* d_out, int out_size, void* d_ws, size_t ws_size,
                              hipStream_t stream) {
  (void)in_sizes; (void)n_in; (void)out_size; (void)ws_size;
  constexpr int B = 4, LQ = 512, LK = 2048, E = 1024, ATT = 1024;

  const float* queries = (const float*)d_in[0];
  const float* context = (const float*)d_in[1];
  const float* Wq  = (const float*)d_in[2];
  const float* bq  = (const float*)d_in[3];
  const float* Wkv = (const float*)d_in[4];
  const float* bkv = (const float*)d_in[5];
  const float* Wo  = (const float*)d_in[6];
  const float* bo  = (const float*)d_in[7];
  float* out = (float*)d_out;

  char* wsp = (char*)d_ws;
  auto carve = [&](size_t bytes) -> char* {
    char* p = wsp; wsp += (bytes + 255) & ~(size_t)255; return p;
  };
  _Float16* qh    = (_Float16*)carve((size_t)B * LQ * E * 2);        //  4 MB
  _Float16* ch    = (_Float16*)carve((size_t)B * LK * E * 2);        // 16 MB
  _Float16* WqT   = (_Float16*)carve((size_t)ATT * E * 2);           //  2 MB
  _Float16* WkvT  = (_Float16*)carve((size_t)2 * ATT * E * 2);       //  4 MB
  _Float16* WoT   = (_Float16*)carve((size_t)E * ATT * 2);           //  2 MB
  _Float16* qproj = (_Float16*)carve((size_t)B * LQ * ATT * 2);      //  4 MB
  _Float16* kvbuf = (_Float16*)carve((size_t)B * LK * 2 * ATT * 2);  // 32 MB
  _Float16* vTb   = (_Float16*)carve((size_t)B * ATT * LK * 2);      // 16 MB
  _Float16* xbuf  = (_Float16*)carve((size_t)B * LQ * ATT * 2);      //  4 MB

  // 1) converts + weight transposes (N-major for B-fragment loads)
  {
    long nq = (long)B * LQ * E, nc = (long)B * LK * E;
    cvt_f32_to_f16_kernel<<<(int)((nq + 255) / 256), 256, 0, stream>>>(queries, qh, nq);
    cvt_f32_to_f16_kernel<<<(int)((nc + 255) / 256), 256, 0, stream>>>(context, ch, nc);
    transpose_cvt_kernel<<<dim3(ATT / 32,     E / 32), dim3(32, 8), 0, stream>>>(Wq,  WqT,  E,   ATT);
    transpose_cvt_kernel<<<dim3(2 * ATT / 32, E / 32), dim3(32, 8), 0, stream>>>(Wkv, WkvT, E,   2 * ATT);
    transpose_cvt_kernel<<<dim3(E / 32,     ATT / 32), dim3(32, 8), 0, stream>>>(Wo,  WoT,  ATT, E);
  }
  // 2) q projection: [2048,1024] x [1024,1024] -> f16
  gemm_wmma_kernel<<<dim3(ATT / 128, (B * LQ) / 128), 256, 0, stream>>>(
      qh, E, WqT, E, bq, qproj, ATT, E, /*cmode=*/1);
  // 3) kv projection: [8192,1024] x [1024,2048] -> f16
  gemm_wmma_kernel<<<dim3((2 * ATT) / 128, (B * LK) / 128), 256, 0, stream>>>(
      ch, E, WkvT, E, bkv, kvbuf, 2 * ATT, E, /*cmode=*/1);
  // 4) V -> V^T
  transpose_v_kernel<<<dim3(LK / 32, ATT / 32, B), dim3(32, 8), 0, stream>>>(kvbuf, vTb);
  // 5) flash attention
  flash_attn_wmma_kernel<<<dim3(LQ / 128, 16, B), 256, 0, stream>>>(qproj, kvbuf, vTb, xbuf);
  // 6) output projection: [2048,1024] x [1024,1024] -> f32
  gemm_wmma_kernel<<<dim3(E / 128, (B * LQ) / 128), 256, 0, stream>>>(
      xbuf, ATT, WoT, ATT, bo, out, E, ATT, /*cmode=*/0);
}
